// CLRHead_60060822667715
// MI455X (gfx1250) — matrix operand order repaired and checked by reading, not compiled, not run on hardware
//
#include <hip/hip_runtime.h>
#include <hip/hip_bf16.h>
#include <stdint.h>

// ---------------------------------------------------------------------------
// CLRNet-style head for MI455X (gfx1250, wave32).  All heavy reductions are
// v_wmma_f32_16x16x32_f16 implicit GEMMs with f32 accumulation.
// ---------------------------------------------------------------------------

typedef __attribute__((ext_vector_type(16))) _Float16 v16h;
typedef __attribute__((ext_vector_type(8)))  float    v8f;

union HF16 {
    v16h     v;
    _Float16 h[16];
    uint4    q[2];
};

#define kB    64
#define kP    192
#define kS    36
#define kC    64
#define kBP   (kB * kP)      // 12288
#define LROWS 58             // LDS tile rows: sp = s + kj reaches 15+9+32 = 56

__device__ __forceinline__ int sample_x(int i) {
    // SAMPLE_X[i] = int(linspace(0,1,36)[i] * 71)
    return (int)((float)i * (1.0f / 35.0f) * 71.0f);
}

static __device__ __forceinline__ v8f zero8() {
    v8f z = {0.f, 0.f, 0.f, 0.f, 0.f, 0.f, 0.f, 0.f};
    return z;
}

// ---------------------------------------------------------------------------
__global__ void cvt_f16_kernel(const float* __restrict__ s, _Float16* __restrict__ d, int n) {
    int i = blockIdx.x * blockDim.x + threadIdx.x;
    if (i < n) d[i] = (_Float16)s[i];
}

// Conv weights (outC, inC, 9) f32  ->  (outC, 9, inC) f16, i.e. K' = kj*inC + c.
// With inC a multiple of 16, every 16-aligned K'-block is contiguous channels
// of a single tap -> B fragments become two b128 LDS loads.
__global__ void cvt_convw_kernel(const float* __restrict__ s, _Float16* __restrict__ d,
                                 int outC, int inC) {
    int i = blockIdx.x * blockDim.x + threadIdx.x;
    int n = outC * inC * 9;
    if (i >= n) return;
    int o  = i / (inC * 9);
    int r  = i - o * inC * 9;
    int c  = r / 9;
    int kj = r - c * 9;
    d[o * inC * 9 + kj * inC + c] = (_Float16)s[i];
}

// priors_b[bp] = priors[p];  pfm[bp][t] = priors[p][6 + SAMPLE_X[t]]
__global__ void init_priors_kernel(const float* __restrict__ priors,
                                   float* __restrict__ priors_b,
                                   float* __restrict__ pfm) {
    int i = blockIdx.x * blockDim.x + threadIdx.x;
    if (i >= kBP) return;
    int p = i % kP;
    for (int k = 0; k < 78; ++k) priors_b[(size_t)i * 78 + k] = priors[p * 78 + k];
    for (int t = 0; t < kS; ++t)
        pfm[(size_t)i * kS + t] = priors[p * 78 + 6 + sample_x(t)];
}

// ---------------------------------------------------------------------------
// Bilinear grid sample -> roi16 (BP, 64, 36) f16.  One block (288 = 8x36
// threads) per (b,p); each thread owns one j and strides channels by 8.
__global__ void grid_sample_kernel(const float* __restrict__ feat, int H, int W,
                                   const float* __restrict__ pfm,
                                   _Float16* __restrict__ roi) {
    const int bp = blockIdx.x;
    const int b  = bp / kP;
    __shared__ float sx[kS], sy[kS];
    if (threadIdx.x < kS) {
        int j  = threadIdx.x;
        int jr = 35 - j;                                     // [:, ::-1]
        float px = pfm[(size_t)bp * kS + jr];
        float fy = 1.0f - (float)sample_x(jr) * (1.0f / 71.0f);  // PRIOR_FEAT_YS[j]
        sx[j] = px * (float)(W - 1);                         // (gx+1)*0.5*(W-1)
        sy[j] = fy * (float)(H - 1);
    }
    __syncthreads();
    const int j  = threadIdx.x % kS;
    const int c0 = threadIdx.x / kS;                         // 0..7
    const float x = sx[j], y = sy[j];
    const float x0 = floorf(x), y0 = floorf(y);
    const float wx = x - x0, wy = y - y0;
    const int ix0 = (int)x0, iy0 = (int)y0;
    float wgt[4];
    int   tap[4];
    bool  vld[4];
#pragma unroll
    for (int t = 0; t < 4; ++t) {
        int ix = ix0 + (t & 1);
        int iy = iy0 + (t >> 1);
        wgt[t] = ((t & 1) ? wx : 1.0f - wx) * ((t >> 1) ? wy : 1.0f - wy);
        vld[t] = (ix >= 0) && (ix < W) && (iy >= 0) && (iy < H);
        tap[t] = min(max(iy, 0), H - 1) * W + min(max(ix, 0), W - 1);
        if (!vld[t]) wgt[t] = 0.0f;
    }
    const float* fb = feat + (size_t)b * kC * H * W;
    for (int c = c0; c < kC; c += 8) {
        const float* fch = fb + (size_t)c * H * W;
        float acc = fch[tap[0]] * wgt[0] + fch[tap[1]] * wgt[1] +
                    fch[tap[2]] * wgt[2] + fch[tap[3]] * wgt[3];
        roi[(size_t)bp * kC * kS + c * kS + j] = (_Float16)acc;
    }
}

// ---------------------------------------------------------------------------
// conv1d (k=9, pad=4) as implicit GEMM with K' = kj*inC + c:
//   out(outC,36) = W'(outC, 9*inC) x im2colT.
// LDS holds the input tile TRANSPOSED: ldsT[sp][c], sp = s+4 in [0,58),
// row stride inC+8 (16B-aligned, bank-skewed), rows sp>=44 zero (covers K tail
// kj==9 and dead columns).  A = two b128 global loads (wave-uniform zero of
// the upper half on the 16-wide K tail).  B = two b128 LDS loads per N-tile;
// the B base index advances incrementally (c0 += 32 with one row-wrap test),
// so the hot loop contains no integer division.
__global__ void conv1d_wmma_kernel(const _Float16* __restrict__ in, int inCtot, int inC,
                                   const _Float16* __restrict__ w,      // (outC, 9, inC)
                                   const float* __restrict__ scale,
                                   const float* __restrict__ shift,
                                   _Float16* __restrict__ out, int outCtot, int oc0, int outC,
                                   int N) {
    extern __shared__ _Float16 lds[];                        // LROWS * (inC+8) halves
    const int tid    = threadIdx.x;
    const int wave   = tid >> 5;
    const int nwaves = blockDim.x >> 5;
    const int lane   = tid & 31;
    const int mtiles = outC >> 4;
    const int Ktot   = inC * 9;
    const int kfull  = Ktot >> 5;                            // complete 32-wide K steps
    const int ktotal = kfull + ((Ktot & 31) ? 1 : 0);        // +1 when remainder (==16)
    const int rstride = inC + 8;                             // LDS row stride (halves)
    const int col    = lane & 15;
    const int bsel   = (lane < 16) ? 0 : 1;                  // A K-suboffset = 16*bsel
    const int boff   = bsel << 4;                            // B K-suboffset

    for (int samp = blockIdx.x; samp < N; samp += gridDim.x) {
        const _Float16* inp = in + (size_t)samp * inCtot * kS;
        for (int sp = wave; sp < LROWS; sp += nwaves) {      // transpose-stage, no divides
            const int s = sp - 4;
            const bool inrow = (s >= 0) && (s < kS);
            for (int c = lane; c < rstride; c += 32) {
                lds[sp * rstride + c] =
                    (inrow && c < inC) ? inp[c * kS + s] : (_Float16)0.0f;
            }
        }
        __syncthreads();
        if (wave < mtiles) {
            v8f acc[3];
            acc[0] = zero8(); acc[1] = zero8(); acc[2] = zero8();
            const int mrow = (wave << 4) + col;              // weight row (A-matrix row)
            const _Float16* wrow = w + (size_t)mrow * Ktot;
            int c0   = boff;                                 // channel within tap kj
            int bidx = col * rstride + boff;                 // = (col+kj)*rstride + c0
            for (int ks = 0; ks < ktotal; ++ks) {
                const int kbase = ks << 5;
                HF16 a;
                const uint4* ap = (const uint4*)(wrow + kbase);
                if (ks != kfull || bsel == 0) {
                    a.q[0] = ap[2 * bsel];                   // k = kbase + 16*bsel + 0..7
                    a.q[1] = ap[2 * bsel + 1];               // k = kbase + 16*bsel + 8..15
                } else {
                    uint4 zq = {0u, 0u, 0u, 0u};
                    a.q[0] = zq;                             // K tail: upper lane half is zero
                    a.q[1] = zq;
                }
#pragma unroll
                for (int nt = 0; nt < 3; ++nt) {
                    HF16 bf;
                    const uint4* bp = (const uint4*)(lds + bidx + nt * 16 * rstride);
                    bf.q[0] = bp[0];
                    bf.q[1] = bp[1];
                    acc[nt] = __builtin_amdgcn_wmma_f32_16x16x32_f16(
                        false, a.v, false, bf.v, (short)0, acc[nt], false, false);
                }
                c0 += 32;                                    // advance K by 32
                bidx += 32;
                if (c0 >= inC) { c0 -= inC; bidx += rstride - inC; }  // next tap row
            }
            _Float16* outp = out + (size_t)samp * outCtot * kS;
#pragma unroll
            for (int nt = 0; nt < 3; ++nt) {
#pragma unroll
                for (int r = 0; r < 8; ++r) {
                    int orow = (wave << 4) + r + ((lane >= 16) ? 8 : 0);
                    int s    = (nt << 4) + col;
                    if (s < kS) {
                        float v = fmaxf(acc[nt][r] * scale[orow] + shift[orow], 0.0f);
                        outp[(oc0 + orow) * kS + s] = (_Float16)v;
                    }
                }
            }
        }
        __syncthreads();
    }
}

// ---------------------------------------------------------------------------
// FC 2304 -> 64 + bias + LayerNorm + ReLU.  A = samples (16 rows/wave),
// B = fc_w^T, K = 2304 (72 WMMA steps), 4 N tiles (h = 0..63).
__global__ void fc_ln_kernel(const _Float16* __restrict__ flat,   // (N, 2304)
                             const _Float16* __restrict__ w,      // (64, 2304)
                             const float* __restrict__ bias,
                             const float* __restrict__ g, const float* __restrict__ bt,
                             float* __restrict__ outf, int N) {
    const int wave  = threadIdx.x >> 5;
    const int lane  = threadIdx.x & 31;
    const int mtile = blockIdx.x * 8 + wave;
    if ((mtile << 4) >= N) return;
    const int bsel = (lane < 16) ? 0 : 1;
    const int h0   = lane & 15;
    const int row  = (mtile << 4) + h0;
    v8f acc[4];
#pragma unroll
    for (int nt = 0; nt < 4; ++nt) acc[nt] = zero8();
    const _Float16* arow = flat + (size_t)row * 2304;
    for (int ks = 0; ks < 72; ++ks) {
        const int kb = ks << 5;
        HF16 a;
        const uint4* ap = (const uint4*)(arow + kb);
        a.q[0] = ap[2 * bsel];                               // k = kb + 16*bsel + 0..7
        a.q[1] = ap[2 * bsel + 1];                           // k = kb + 16*bsel + 8..15
#pragma unroll
        for (int nt = 0; nt < 4; ++nt) {
            HF16 bf;
            const uint4* bp =
                (const uint4*)(w + (size_t)((nt << 4) + h0) * 2304 + kb + (bsel << 4));
            bf.q[0] = bp[0];
            bf.q[1] = bp[1];
            acc[nt] = __builtin_amdgcn_wmma_f32_16x16x32_f16(
                false, a.v, false, bf.v, (short)0, acc[nt], false, false);
        }
    }
    // bias + LayerNorm(64) + ReLU, per D-matrix row via shfl_xor within 16-lane halves
#pragma unroll
    for (int r = 0; r < 8; ++r) {
        float v[4];
        float s = 0.0f, sq = 0.0f;
#pragma unroll
        for (int nt = 0; nt < 4; ++nt) {
            v[nt] = acc[nt][r] + bias[(nt << 4) + h0];
            s  += v[nt];
            sq += v[nt] * v[nt];
        }
#pragma unroll
        for (int m = 1; m < 16; m <<= 1) {
            s  += __shfl_xor(s,  m, 32);
            sq += __shfl_xor(sq, m, 32);
        }
        const float mu  = s * (1.0f / 64.0f);
        const float var = sq * (1.0f / 64.0f) - mu * mu;
        const float inv = rsqrtf(var + 1e-5f);
        const int srow  = (mtile << 4) + r + ((lane >= 16) ? 8 : 0);
#pragma unroll
        for (int nt = 0; nt < 4; ++nt) {
            const int h = (nt << 4) + h0;
            outf[(size_t)srow * 64 + h] = fmaxf((v[nt] - mu) * inv * g[h] + bt[h], 0.0f);
        }
    }
}

// ---------------------------------------------------------------------------
// value/key 1x1 convs evaluated ONLY at the 250 resize_nearest(10,25) pixels.
__global__ void valkey_kernel(const float* __restrict__ feat, int H, int W,
                              const float* __restrict__ vw, const float* __restrict__ vb,
                              const float* __restrict__ kw, const float* __restrict__ ksc,
                              const float* __restrict__ ksh,
                              float* __restrict__ value, float* __restrict__ key) {
    const int b = blockIdx.x;
    __shared__ int sidx[250];
    for (int t = threadIdx.x; t < 250; t += blockDim.x) {
        int iy = ((t / 25) * H) / 10;
        int ix = ((t % 25) * W) / 25;
        sidx[t] = iy * W + ix;
    }
    __syncthreads();
    const float* fb = feat + (size_t)b * kC * H * W;
    for (int ot = threadIdx.x; ot < 64 * 250; ot += blockDim.x) {
        int o = ot / 250, t = ot - o * 250;
        const float* fp  = fb + sidx[t];
        const float* wvo = vw + o * 64;
        const float* wko = kw + o * 64;
        float sv = 0.0f, sk = 0.0f;
        for (int c = 0; c < 64; ++c) {
            float f = fp[(size_t)c * H * W];
            sv += f * wvo[c];
            sk += f * wko[c];
        }
        value[((size_t)b * 64 + o) * 250 + t] = sv + vb[o];
        key[((size_t)b * 64 + o) * 250 + t]   = fmaxf(sk * ksc[o] + ksh[o], 0.0f);
    }
}

// ---------------------------------------------------------------------------
// Attention: per (b,p): 250 scores, softmax, context, residual.  One block each.
__global__ void attn_kernel(const float* __restrict__ roi_fc,
                            const float* __restrict__ key, const float* __restrict__ value,
                            const float* __restrict__ fqw, const float* __restrict__ fqb,
                            const float* __restrict__ aw, const float* __restrict__ ab,
                            float* __restrict__ fc_feat, _Float16* __restrict__ fc_feat16) {
    const int bp  = blockIdx.x;
    const int b   = bp / kP, p = bp % kP;
    const int tid = threadIdx.x;
    __shared__ float q[64];
    __shared__ float sim[256];
    __shared__ float red[256];
    __shared__ float ctxp[4][64];
    if (tid < 64) q[tid] = fmaxf(roi_fc[(size_t)bp * 64 + tid] * fqw[p] + fqb[p], 0.0f);
    __syncthreads();
    float sc = -1e30f;
    if (tid < 250) {
        const float* kp = key + (size_t)b * 64 * 250 + tid;
        float d = 0.0f;
        for (int c = 0; c < 64; ++c) d += q[c] * kp[c * 250];
        sc = d * 0.125f;                                     // C^-0.5 = 1/8
    }
    red[tid] = sc;
    __syncthreads();
    for (int st = 128; st > 0; st >>= 1) {
        if (tid < st) red[tid] = fmaxf(red[tid], red[tid + st]);
        __syncthreads();
    }
    float mx = red[0];
    __syncthreads();
    float e = (tid < 250) ? __expf(sc - mx) : 0.0f;
    red[tid] = e;
    __syncthreads();
    for (int st = 128; st > 0; st >>= 1) {
        if (tid < st) red[tid] += red[tid + st];
        __syncthreads();
    }
    float sm = red[0];
    sim[tid] = e / sm;
    __syncthreads();
    const int o = tid & 63, qd = tid >> 6;
    const float* vp = value + (size_t)b * 64 * 250 + (size_t)o * 250;
    float a = 0.0f;
    for (int k = qd; k < 250; k += 4) a += sim[k] * vp[k];
    ctxp[qd][o] = a;
    __syncthreads();
    if (tid < 64) {
        float ctx = ctxp[0][tid] + ctxp[1][tid] + ctxp[2][tid] + ctxp[3][tid];
        float ro  = roi_fc[(size_t)bp * 64 + tid] + ctx * aw[p] + ab[p];
        fc_feat[(size_t)bp * 64 + tid]   = ro;
        fc_feat16[(size_t)bp * 64 + tid] = (_Float16)ro;
    }
}

// ---------------------------------------------------------------------------
// Two-layer 64x64 MLP (+ReLU) via WMMA; inter-layer D->A re-layout through a
// per-wave LDS transpose tile.
__global__ void mlp2_kernel(const _Float16* __restrict__ inF,   // (N, 64)
                            const _Float16* __restrict__ w,     // (2, 64, 64)
                            const float* __restrict__ bias,     // (2, 64)
                            float* __restrict__ outF, int N) {
    __shared__ __align__(16) _Float16 tile[8][16 * 64];
    const int wave  = threadIdx.x >> 5;
    const int lane  = threadIdx.x & 31;
    const int mtile = blockIdx.x * 8 + wave;
    const int bsel  = (lane < 16) ? 0 : 1;
    const int h0    = lane & 15;
    const int row   = (mtile << 4) + h0;
    _Float16* myt   = tile[wave];
#pragma unroll
    for (int l = 0; l < 2; ++l) {
        v8f acc[4];
#pragma unroll
        for (int nt = 0; nt < 4; ++nt) acc[nt] = zero8();
        const _Float16* wl = w + l * 64 * 64;
#pragma unroll
        for (int ks = 0; ks < 2; ++ks) {
            const int kb = ks << 5;
            HF16 a;
            const uint4* ap = (l == 0) ? (const uint4*)(inF + (size_t)row * 64 + kb)
                                       : (const uint4*)(myt + h0 * 64 + kb);
            a.q[0] = ap[2 * bsel];
            a.q[1] = ap[2 * bsel + 1];
#pragma unroll
            for (int nt = 0; nt < 4; ++nt) {
                HF16 bf;
                const uint4* bp = (const uint4*)(wl + ((nt << 4) + h0) * 64 + kb + (bsel << 4));
                bf.q[0] = bp[0];
                bf.q[1] = bp[1];
                acc[nt] = __builtin_amdgcn_wmma_f32_16x16x32_f16(
                    false, a.v, false, bf.v, (short)0, acc[nt], false, false);
            }
        }
#pragma unroll
        for (int nt = 0; nt < 4; ++nt) {
#pragma unroll
            for (int r = 0; r < 8; ++r) {
                int rr = r + ((lane >= 16) ? 8 : 0);
                int h  = (nt << 4) + h0;
                float v = fmaxf(acc[nt][r] + bias[l * 64 + h], 0.0f);
                if (l == 0) myt[rr * 64 + h] = (_Float16)v;
                else        outF[(size_t)((mtile << 4) + rr) * 64 + h] = v;
            }
        }
        __syncthreads();
    }
}

// ---------------------------------------------------------------------------
// Heads (2 & 76 dots) + geometric refinement; writes preds and updates priors.
// off(j) is affine in prior_y:  off = offA - py*offB  (no local arrays/scratch).
__global__ void heads_preds_kernel(const float* __restrict__ clsf,
                                   const float* __restrict__ regf,
                                   const float* __restrict__ chw, const float* __restrict__ chb,
                                   const float* __restrict__ rhw, const float* __restrict__ rhb,
                                   float* __restrict__ priors_b, float* __restrict__ pfm,
                                   float* __restrict__ out, int stage) {
    int i = blockIdx.x * blockDim.x + threadIdx.x;
    if (i >= kBP) return;
    const float* cf = clsf + (size_t)i * 64;
    const float* rf = regf + (size_t)i * 64;
    float c0 = chb[0], c1 = chb[1];
    for (int c = 0; c < 64; ++c) { c0 += cf[c] * chw[c]; c1 += cf[c] * chw[64 + c]; }
    float r03[4];
#pragma unroll
    for (int o = 0; o < 4; ++o) {
        float a = rhb[o];
        const float* wr = rhw + o * 64;
        for (int c = 0; c < 64; ++c) a += rf[c] * wr[c];
        r03[o] = a;
    }
    float* pb = priors_b + (size_t)i * 78;
    float p0 = pb[2] + r03[0];
    float p1 = pb[3] + r03[1];
    float p2 = pb[4] + r03[2];
    float p5 = r03[3];
    float invt = 1.0f / tanf(p2 * 3.14159265358979323846f + 1e-5f);
    float offB = (512.0f * invt) * (1.0f / 639.0f);
    float offA = (p1 * 639.0f + (1.0f - p0) * 512.0f * invt) * (1.0f / 639.0f);
    float* orow = out + (size_t)i * 78;
    orow[0] = c0; orow[1] = c1; orow[2] = p0; orow[3] = p1; orow[4] = p2; orow[5] = p5;
    for (int j = 0; j < 72; ++j) {
        float rgj = rhb[4 + j];
        const float* wr = rhw + (4 + j) * 64;
        for (int c = 0; c < 64; ++c) rgj += rf[c] * wr[c];
        float py = 1.0f - (float)j * (1.0f / 71.0f);         // PRIOR_YS[j]
        orow[6 + j] = (offA - py * offB) + rgj;
    }
    if (stage < 2) {
        pb[0] = c0; pb[1] = c1; pb[2] = p0; pb[3] = p1; pb[4] = p2; pb[5] = p5;
        for (int j = 0; j < 72; ++j) {
            float py = 1.0f - (float)j * (1.0f / 71.0f);
            pb[6 + j] = offA - py * offB;
        }
        for (int t = 0; t < kS; ++t) {
            float py = 1.0f - (float)sample_x(t) * (1.0f / 71.0f);
            pfm[(size_t)i * kS + t] = offA - py * offB;
        }
    }
}

// ---------------------------------------------------------------------------
extern "C" void kernel_launch(void* const* d_in, const int* in_sizes, int n_in,
                              void* d_out, int out_size, void* d_ws, size_t ws_size,
                              hipStream_t stream) {
    (void)in_sizes; (void)n_in; (void)out_size; (void)ws_size;
    const float* feat[3]   = {(const float*)d_in[0], (const float*)d_in[1], (const float*)d_in[2]};
    const int Hs[3] = {64, 32, 16}, Ws[3] = {80, 40, 20};
    const float* priors      = (const float*)d_in[3];
    const float* convs_w     = (const float*)d_in[4];
    const float* convs_scale = (const float*)d_in[5];
    const float* convs_shift = (const float*)d_in[6];
    const float* cat_w[3]    = {(const float*)d_in[7], (const float*)d_in[8], (const float*)d_in[9]};
    const float* cat_scale   = (const float*)d_in[10];
    const float* cat_shift   = (const float*)d_in[11];
    const float* fkey_w      = (const float*)d_in[12];
    const float* fkey_scale  = (const float*)d_in[13];
    const float* fkey_shift  = (const float*)d_in[14];
    const float* fval_w      = (const float*)d_in[15];
    const float* fval_b      = (const float*)d_in[16];
    const float* fq_w        = (const float*)d_in[17];
    const float* fq_b        = (const float*)d_in[18];
    const float* attW_w      = (const float*)d_in[19];
    const float* attW_b      = (const float*)d_in[20];
    const float* fc_w        = (const float*)d_in[21];
    const float* fc_b        = (const float*)d_in[22];
    const float* ln_g        = (const float*)d_in[23];
    const float* ln_b        = (const float*)d_in[24];
    const float* cls_mlp_w   = (const float*)d_in[25];
    const float* cls_mlp_b   = (const float*)d_in[26];
    const float* reg_mlp_w   = (const float*)d_in[27];
    const float* reg_mlp_b   = (const float*)d_in[28];
    const float* cls_head_w  = (const float*)d_in[29];
    const float* cls_head_b  = (const float*)d_in[30];
    const float* reg_head_w  = (const float*)d_in[31];
    const float* reg_head_b  = (const float*)d_in[32];
    float* out = (float*)d_out;

    char* base = (char*)d_ws;
    size_t off = 0;
    auto alloc = [&](size_t bytes) -> void* {
        void* pp = base + off;
        off = (off + bytes + 255) & ~(size_t)255;
        return pp;
    };
    float*    priors_b  = (float*)alloc((size_t)kBP * 78 * 4);
    float*    pfm       = (float*)alloc((size_t)kBP * kS * 4);
    _Float16* roi16     = (_Float16*)alloc((size_t)kBP * kC * kS * 2);
    _Float16* cf16      = (_Float16*)alloc((size_t)kBP * 144 * kS * 2);   // cached across stages
    _Float16* cat16     = (_Float16*)alloc((size_t)kBP * kC * kS * 2);
    float*    roi_fc    = (float*)alloc((size_t)kBP * 64 * 4);
    float*    valbuf    = (float*)alloc((size_t)kB * 64 * 250 * 4);
    float*    keybuf    = (float*)alloc((size_t)kB * 64 * 250 * 4);
    float*    fc_feat   = (float*)alloc((size_t)kBP * 64 * 4);
    _Float16* fc_feat16 = (_Float16*)alloc((size_t)kBP * 64 * 2);
    float*    clsf2     = (float*)alloc((size_t)kBP * 64 * 4);
    float*    regf2     = (float*)alloc((size_t)kBP * 64 * 4);
    _Float16* convs_w16 = (_Float16*)alloc((size_t)3 * 48 * 576 * 2);
    _Float16* cat_w16_0 = (_Float16*)alloc((size_t)64 * 432 * 2);
    _Float16* cat_w16_1 = (_Float16*)alloc((size_t)64 * 864 * 2);
    _Float16* cat_w16_2 = (_Float16*)alloc((size_t)64 * 1296 * 2);
    _Float16* cat_w16[3] = {cat_w16_0, cat_w16_1, cat_w16_2};
    _Float16* fc_w16    = (_Float16*)alloc((size_t)64 * 2304 * 2);
    _Float16* cls_mlp16 = (_Float16*)alloc((size_t)2 * 64 * 64 * 2);
    _Float16* reg_mlp16 = (_Float16*)alloc((size_t)2 * 64 * 64 * 2);

    auto cvt = [&](const float* s, _Float16* d, int n) {
        cvt_f16_kernel<<<(n + 255) / 256, 256, 0, stream>>>(s, d, n);
    };
    // conv weights: reorder (outC, inC, 9) -> (outC, 9, inC).  The three
    // per-scale convs share inC=64, so convert them as one outC=144 batch.
    cvt_convw_kernel<<<(3 * 48 * 576 + 255) / 256, 256, 0, stream>>>(
        convs_w, convs_w16, 3 * 48, 64);
    for (int st = 0; st < 3; ++st) {
        int n = 64 * 48 * (st + 1) * 9;
        cvt_convw_kernel<<<(n + 255) / 256, 256, 0, stream>>>(
            cat_w[st], cat_w16[st], 64, 48 * (st + 1));
    }
    cvt(fc_w,      fc_w16, 64 * 2304);
    cvt(cls_mlp_w, cls_mlp16, 2 * 64 * 64);
    cvt(reg_mlp_w, reg_mlp16, 2 * 64 * 64);

    init_priors_kernel<<<(kBP + 255) / 256, 256, 0, stream>>>(priors, priors_b, pfm);

    for (int st = 0; st < 3; ++st) {
        grid_sample_kernel<<<kBP, 288, 0, stream>>>(feat[st], Hs[st], Ws[st], pfm, roi16);
        // per-scale conv (only the new one; cf[i<st] cached from earlier stages)
        conv1d_wmma_kernel<<<kBP, 128, LROWS * (64 + 8) * 2, stream>>>(
            roi16, 64, 64, convs_w16 + (size_t)st * 48 * 576,
            convs_scale + st * 48, convs_shift + st * 48,
            cf16, 144, 48 * st, 48, kBP);
        // concat conv over the first 48*(st+1) channels of cf16
        const int inC = 48 * (st + 1);
        conv1d_wmma_kernel<<<kBP, 128, LROWS * (inC + 8) * 2, stream>>>(
            cf16, 144, inC, cat_w16[st],
            cat_scale + st * 64, cat_shift + st * 64,
            cat16, 64, 0, 64, kBP);
        fc_ln_kernel<<<kBP / 128, 256, 0, stream>>>(cat16, fc_w16, fc_b, ln_g, ln_b,
                                                    roi_fc, kBP);
        valkey_kernel<<<kB, 256, 0, stream>>>(feat[st], Hs[st], Ws[st],
                                              fval_w, fval_b, fkey_w, fkey_scale, fkey_shift,
                                              valbuf, keybuf);
        attn_kernel<<<kBP, 256, 0, stream>>>(roi_fc, keybuf, valbuf, fq_w, fq_b,
                                             attW_w, attW_b, fc_feat, fc_feat16);
        mlp2_kernel<<<kBP / 128, 256, 0, stream>>>(fc_feat16, cls_mlp16, cls_mlp_b, clsf2, kBP);
        mlp2_kernel<<<kBP / 128, 256, 0, stream>>>(fc_feat16, reg_mlp16, reg_mlp_b, regf2, kBP);
        heads_preds_kernel<<<(kBP + 255) / 256, 256, 0, stream>>>(
            clsf2, regf2, cls_head_w, cls_head_b, reg_head_w, reg_head_b,
            priors_b, pfm, out + (size_t)st * kBP * 78, st);
    }
}
